// CombinedQLSTM_65481071400329
// MI455X (gfx1250) — matrix operand order repaired and checked by reading, not compiled, not run on hardware
//
#include <hip/hip_runtime.h>
#include <cstdint>

typedef unsigned short u16;
typedef __attribute__((ext_vector_type(16))) __bf16 v16bf;
typedef __attribute__((ext_vector_type(8)))  float  v8f;
typedef __attribute__((ext_vector_type(8)))  unsigned short us8;
typedef __attribute__((ext_vector_type(4)))  unsigned int   uv4;
typedef __attribute__((ext_vector_type(8)))  int            iv8;
typedef __attribute__((ext_vector_type(4)))  int            iv4;

#define TT 1024
#define BB 128
#define DD 256
#define HH 256
#define DH 512
#define G4H 1024

#define LDA_STRIDE 520      // ushorts per A row (16 rows x 512 K, bank padded)
#define LDB_STRIDE 40       // ushorts per B row (32 K + TDM pad 4 dwords -> 80B)
#define BSLAB_BYTES (256 * LDB_STRIDE * 2)   // 20480 bytes per K-slab

#if __has_builtin(__builtin_amdgcn_tensor_load_to_lds) && __has_builtin(__builtin_amdgcn_s_wait_tensorcnt)
#define USE_TDM 1
#else
#define USE_TDM 0
#endif

__device__ __forceinline__ u16 f2bf(float f) {
  unsigned u = __builtin_bit_cast(unsigned, f);
  u += 0x7FFFu + ((u >> 16) & 1u);
  return (u16)(u >> 16);
}
__device__ __forceinline__ float sigm(float x) { return 1.f / (1.f + __expf(-x)); }

union FragU { us8 u[2]; v16bf v; };

#if USE_TDM
// Issue one TDM 2-D tile load: 256 weight rows x 32 bf16 (one K-slab) into LDS
// at byte offset `lds`, with LDS padding of 4 dwords per 16-dword row (80B row
// stride) so WMMA B-fragment reads spread across banks.
__device__ __forceinline__ void tdm_load_slab(const u16* gsrc, unsigned lds) {
  unsigned long long ga = (unsigned long long)(const void*)gsrc;
  uv4 g0;
  g0.x = 1u;                                        // count=1, user mode
  g0.y = lds;                                       // lds_addr (bytes)
  g0.z = (unsigned)(ga & 0xFFFFFFFFull);            // global_addr[31:0]
  g0.w = (unsigned)((ga >> 32) & 0x1FFFFFFull) | (2u << 30); // addr[56:32]|type=2
  iv8 g1;
  g1[0] = (1 << 16)            // data_size = 2B
        | (1 << 20)            // pad_enable
        | (3 << 22)            // pad_interval: 16 dwords (one 32-elem bf16 row)
        | (3 << 25);           // pad_amount: 4 dwords -> 80B LDS row stride
  g1[1] = (DH & 0xFFFF) << 16; // tensor_dim0 = 512 (low16)
  g1[2] = (G4H & 0xFFFF) << 16;// dim0 hi=0 | tensor_dim1 = 1024 (low16)
  g1[3] = (32 << 16);          // dim1 hi=0 | tile_dim0 = 32
  g1[4] = 256;                 // tile_dim1 = 256, tile_dim2 = 0
  g1[5] = DH;                  // tensor_dim0_stride = 512 (low32)
  g1[6] = (DH & 0xFFFF) << 16; // stride0 hi=0 | tensor_dim1_stride low16 = 512
  g1[7] = 0;                   // tensor_dim1_stride hi
  iv4 g2 = {0, 0, 0, 0}, g3 = {0, 0, 0, 0};
  iv8 gx = {0, 0, 0, 0, 0, 0, 0, 0};
  __builtin_amdgcn_tensor_load_to_lds(g0, g1, g2, g3, gx, 0);
}
#endif

// ---------------------------------------------------------------------------
// Kernel 0: weights fp32->bf16 (rows stacked f,i,g,o), stacked biases, zero
// h/c state.
// ---------------------------------------------------------------------------
__global__ __launch_bounds__(256) void setup_kernel(
    const float* __restrict__ Wf, const float* __restrict__ bfv,
    const float* __restrict__ Wi, const float* __restrict__ biv,
    const float* __restrict__ Wg, const float* __restrict__ bgv,
    const float* __restrict__ Wo, const float* __restrict__ bov,
    u16* __restrict__ Wbf, float* __restrict__ biasAll,
    float* __restrict__ hbuf, float* __restrict__ cbuf) {
  const float* Ws[4] = {Wf, Wi, Wg, Wo};
  const float* bs[4] = {bfv, biv, bgv, bov};
  int tid = blockIdx.x * blockDim.x + threadIdx.x;      // 131072 threads
  for (int idx = tid; idx < 4 * HH * DH; idx += 131072) {
    int gate = idx >> 17;                               // 256*512 per gate
    int rem  = idx & 131071;
    Wbf[idx] = f2bf(Ws[gate][rem]);
  }
  if (tid < G4H) biasAll[tid] = bs[tid >> 8][tid & 255];
  if (tid < BB * HH) { hbuf[tid] = 0.f; cbuf[tid] = 0.f; }
}

// ---------------------------------------------------------------------------
// Kernel 1 (per step): gates[B,4H] = act( [x_t , h] @ W^T + b )
// grid = 32 blocks (8 M-tiles x 4 N-quadrants), 256 threads = 8 waves.
// A (16x512 bf16) staged once in LDS; B K-slabs (256x32 bf16) double-buffered
// via the Tensor Data Mover so each DMA overlaps the previous slab's WMMAs.
// ---------------------------------------------------------------------------
__global__ __launch_bounds__(256) void gate_gemm_kernel(
    const float* __restrict__ xall, const u16* __restrict__ Wbf,
    const float* __restrict__ biasAll, const float* __restrict__ hbuf,
    float* __restrict__ gbuf, int t) {
  __shared__ u16 As[16 * LDA_STRIDE];
  __shared__ u16 Bs[2 * 256 * LDB_STRIDE];

  const int mt = blockIdx.x >> 2, nq = blockIdx.x & 3;
  const int tid = threadIdx.x, wave = tid >> 5, lane = tid & 31;
  const int lh = lane >> 4, al = lane & 15;
  const int m0 = mt * 16;
  const float* xrow = xall + (size_t)t * BB * DD;
  const u16* wbase = Wbf + (size_t)nq * 256 * DH;

  // stage comb=[x,h] tile, converting fp32 -> bf16 into LDS
  {
    int r  = tid >> 4;
    int c0 = (tid & 15) * 32;
    const float* src = (c0 < DD) ? (xrow + (size_t)(m0 + r) * DD + c0)
                                 : (hbuf + (size_t)(m0 + r) * HH + (c0 - DD));
    for (int e = 0; e < 32; e += 2) {
      unsigned pack = (unsigned)f2bf(src[e]) | ((unsigned)f2bf(src[e + 1]) << 16);
      *(unsigned*)&As[r * LDA_STRIDE + c0 + e] = pack;
    }
  }

#if USE_TDM
  const unsigned ldsB = (unsigned)(uintptr_t)(void*)Bs;  // low 32 bits = LDS off
  if (wave == 0) tdm_load_slab(wbase, ldsB);             // prologue: slab 0
#else
  {
    int r = tid;
    const u16* src = wbase + (size_t)r * DH;
    for (int i = 0; i < 4; ++i)
      *(us8*)&Bs[r * LDB_STRIDE + i * 8] = *(const us8*)&src[i * 8];
  }
#endif

  const int nw0 = nq * 256 + wave * 32;
  v8f acc0 = {}, acc1 = {};

  for (int kb = 0; kb < 16; ++kb) {
    const int buf = kb & 1;
#if USE_TDM
    if (wave == 0) __builtin_amdgcn_s_wait_tensorcnt(0);   // slab kb landed
#endif
    __syncthreads();   // publish slab kb (+ A on first iter); old slab free
#if USE_TDM
    if (kb + 1 < 16 && wave == 0)                          // overlap next DMA
      tdm_load_slab(wbase + (kb + 1) * 32, ldsB + (buf ^ 1) * BSLAB_BYTES);
#else
    if (kb + 1 < 16) {
      int r = tid;
      const u16* src = wbase + (size_t)r * DH + (kb + 1) * 32;
      u16* dst = &Bs[(buf ^ 1) * 256 * LDB_STRIDE + r * LDB_STRIDE];
      for (int i = 0; i < 4; ++i)
        *(us8*)&dst[i * 8] = *(const us8*)&src[i * 8];
    }
#endif
    if (kb + 2 < 16)   // warm GL2 for the slab after next
      __builtin_prefetch(&wbase[(size_t)tid * DH + (kb + 2) * 32], 0, 0);

    // A fragment: row = al, 16 consecutive K at 16*lh
    FragU af, bfr0, bfr1;
    int abase = al * LDA_STRIDE + kb * 32 + lh * 16;
    af.u[0] = *(const us8*)&As[abase];
    af.u[1] = *(const us8*)&As[abase + 8];
    // B fragments: column = al, 16 consecutive K at 16*lh
    const u16* bslab = &Bs[buf * 256 * LDB_STRIDE];
    int b0 = (wave * 32 + al) * LDB_STRIDE + lh * 16;
    bfr0.u[0] = *(const us8*)&bslab[b0];
    bfr0.u[1] = *(const us8*)&bslab[b0 + 8];
    int b1 = (wave * 32 + 16 + al) * LDB_STRIDE + lh * 16;
    bfr1.u[0] = *(const us8*)&bslab[b1];
    bfr1.u[1] = *(const us8*)&bslab[b1 + 8];

    acc0 = __builtin_amdgcn_wmma_f32_16x16x32_bf16(false, af.v, false, bfr0.v,
                                                   (short)0, acc0, false, false);
    acc1 = __builtin_amdgcn_wmma_f32_16x16x32_bf16(false, af.v, false, bfr1.v,
                                                   (short)0, acc1, false, false);
    __syncthreads();   // all reads of slab kb done before its buffer is reused
  }

  // epilogue: bias + activation (g gate = tanh, others sigmoid)
  for (int r = 0; r < 8; ++r) {
    int m = m0 + lh * 8 + r;
    int col0 = nw0 + al;
    float v0 = acc0[r] + biasAll[col0];
    gbuf[(size_t)m * G4H + col0] = ((col0 >> 8) == 2) ? tanhf(v0) : sigm(v0);
    int col1 = nw0 + 16 + al;
    float v1 = acc1[r] + biasAll[col1];
    gbuf[(size_t)m * G4H + col1] = ((col1 >> 8) == 2) ? tanhf(v1) : sigm(v1);
  }
}

// ---------------------------------------------------------------------------
// Kernel 2 (per step): estimator MLPs + cell/hidden update.
// grid = 128 blocks (one per batch row), 256 threads = 8 waves.
// ---------------------------------------------------------------------------
__global__ __launch_bounds__(256) void est_update_kernel(
    const float* __restrict__ gbuf,
    const float* __restrict__ estW1, const float* __restrict__ estb1,
    const float* __restrict__ estW2, const float* __restrict__ estb2,
    const float* __restrict__ estW3, const float* __restrict__ estb3,
    float* __restrict__ hbuf, float* __restrict__ cbuf,
    float* __restrict__ out, int t) {
  __shared__ float grow[G4H];
  __shared__ float h1s[32];
  __shared__ float h2s[16];
  __shared__ float estv[4];

  const int b = blockIdx.x, tid = threadIdx.x;
  const int wave = tid >> 5, lane = tid & 31;

  *(float4*)&grow[tid * 4] = *(const float4*)&gbuf[(size_t)b * G4H + tid * 4];
  __syncthreads();

  // layer 1: 32 dot products (4 gates x 8 units) of length 256, 4 per wave
  for (int p = wave * 4; p < wave * 4 + 4; ++p) {
    int k = p >> 3;
    const float* w = estW1 + p * HH;
    const float* g = grow + k * HH;
    float s = 0.f;
    for (int u = lane; u < HH; u += 32) s += g[u] * w[u];
    for (int off = 16; off; off >>= 1) s += __shfl_xor(s, off, 32);
    if (lane == 0) h1s[p] = tanhf(s + estb1[p]);
  }
  __syncthreads();

  if (tid < 16) {   // layer 2: 4 gates x 4 units, dot length 8
    int k = tid >> 2, u = tid & 3;
    float s = estb2[k * 4 + u];
    for (int j = 0; j < 8; ++j) s += h1s[k * 8 + j] * estW2[(k * 4 + u) * 8 + j];
    h2s[tid] = tanhf(s);
  }
  __syncthreads();

  if (tid < 4) {    // layer 3: sigmoid head per gate
    float s = estb3[tid];
    for (int u = 0; u < 4; ++u) s += h2s[tid * 4 + u] * estW3[tid * 4 + u];
    estv[tid] = sigm(s);
  }
  __syncthreads();

  // cell update: tid = hidden index
  float f = grow[tid]           * estv[0];
  float i = grow[HH + tid]      * estv[1];
  float g = grow[2 * HH + tid]  * estv[2];
  float o = grow[3 * HH + tid]  * estv[3];
  float c = f * cbuf[(size_t)b * HH + tid] + i * g;
  float h = o * tanhf(c);
  cbuf[(size_t)b * HH + tid] = c;
  hbuf[(size_t)b * HH + tid] = h;
  out[(size_t)t * BB * HH + (size_t)b * HH + tid] = h;
  if (t == TT - 1) {
    size_t base = (size_t)TT * BB * HH;
    out[base + (size_t)b * HH + tid] = h;                   // hx
    out[base + (size_t)BB * HH + (size_t)b * HH + tid] = c; // cx
  }
}

// ---------------------------------------------------------------------------
extern "C" void kernel_launch(void* const* d_in, const int* in_sizes, int n_in,
                              void* d_out, int out_size, void* d_ws, size_t ws_size,
                              hipStream_t stream) {
  const float* inputs = (const float*)d_in[0];
  const float* Wf = (const float*)d_in[1];  const float* bfv = (const float*)d_in[2];
  const float* Wi = (const float*)d_in[3];  const float* biv = (const float*)d_in[4];
  const float* Wg = (const float*)d_in[5];  const float* bgv = (const float*)d_in[6];
  const float* Wo = (const float*)d_in[7];  const float* bov = (const float*)d_in[8];
  const float* estW1 = (const float*)d_in[9];   const float* estb1 = (const float*)d_in[10];
  const float* estW2 = (const float*)d_in[11];  const float* estb2 = (const float*)d_in[12];
  const float* estW3 = (const float*)d_in[13];  const float* estb3 = (const float*)d_in[14];
  float* out = (float*)d_out;

  char* ws = (char*)d_ws;
  u16*   Wbf     = (u16*)ws;                       // 4*256*512 bf16 = 1 MB
  float* biasAll = (float*)(ws + 1048576);         // 1024 f32
  float* gbuf    = (float*)(ws + 1052672);         // 128*1024 f32
  float* hbuf    = (float*)(ws + 1576960);         // 128*256 f32
  float* cbuf    = (float*)(ws + 1708032);         // 128*256 f32

  setup_kernel<<<512, 256, 0, stream>>>(Wf, bfv, Wi, biv, Wg, bgv, Wo, bov,
                                        Wbf, biasAll, hbuf, cbuf);
  for (int t = 0; t < TT; ++t) {
    gate_gemm_kernel<<<32, 256, 0, stream>>>(inputs, Wbf, biasAll, hbuf, gbuf, t);
    est_update_kernel<<<128, 256, 0, stream>>>(gbuf, estW1, estb1, estW2, estb2,
                                               estW3, estb3, hbuf, cbuf, out, t);
  }
}
